// GraphCNN_27702539059361
// MI455X (gfx1250) — matrix operand order repaired and checked by reading, not compiled, not run on hardware
//
#include <hip/hip_runtime.h>
#include <hip/hip_bf16.h>

#define N_NODES 50000
#define N_EDGES 800000
#define N_GRAPH 256
#define DIN     200
#define KP0     224            // DIN padded to multiple of 32
#define DH      128
#define OUTC    (DIN + 4*DH)   // 712
#define BN_EPS  1e-5f

typedef __attribute__((ext_vector_type(16))) __bf16 bf16x16;
typedef __attribute__((ext_vector_type(8)))  float  f32x8;

union Frag16 {
    bf16x16      v;
    unsigned int w[8];
    uint4        q[2];
};

// split two f32 into packed bf16-hi word and bf16-lo (residual) word
__device__ __forceinline__ void split2(float f0, float f1,
                                       unsigned int& hw, unsigned int& lw) {
    unsigned int i0 = __float_as_uint(f0);
    unsigned int i1 = __float_as_uint(f1);
    unsigned int h0 = i0 >> 16;
    unsigned int h1 = i1 & 0xffff0000u;
    float r0 = f0 - __uint_as_float(h0 << 16);
    float r1 = f1 - __uint_as_float(h1);
    hw = h1 | h0;
    lw = (__float_as_uint(r1) & 0xffff0000u) | (__float_as_uint(r0) >> 16);
}

// ---- pooled = (1+eps)*h, one block per row, zero-fills K padding ----
__global__ void k_scale_pad(const float* __restrict__ h, const float* __restrict__ eps,
                            float* __restrict__ P, int D, int ldp) {
    int row = blockIdx.x;
    int col = threadIdx.x;
    float s = 1.0f + *eps;
    if (col < D)
        P[(size_t)row * ldp + col] = s * h[(size_t)row * D + col];
    else if (col < ldp)
        P[(size_t)row * ldp + col] = 0.0f;
}

// ---- pooled[src] += h[dst], one wave per edge ----
__global__ void k_scatter(const int* __restrict__ ei, const float* __restrict__ h,
                          float* __restrict__ pooled, int E, int D, int ldp) {
    int wid  = (blockIdx.x * blockDim.x + threadIdx.x) >> 5;
    int lane = threadIdx.x & 31;
    if (wid >= E) return;
    int s = ei[wid];
    int d = ei[E + wid];
    const float* hrow = h + (size_t)d * D;
    float*       prow = pooled + (size_t)s * ldp;
    for (int c = lane; c < D; c += 32)
        atomicAdd(&prow[c], hrow[c]);
}

// ---- pre-split + pre-swizzle W[kreal x 128] into per-lane WMMA B fragments ----
// Layout: group g = (t*8 + colTile)*32 + lane owns 16 dwords:
//         [0..7] = bf16-hi packed words, [8..15] = bf16-lo packed words.
template <int K>
__global__ void k_prepw(const float* __restrict__ W, int kreal,
                        unsigned int* __restrict__ Wf) {
    constexpr int NK = K / 32;
    int idx = blockIdx.x * blockDim.x + threadIdx.x;   // one thread per (g, j)
    if (idx >= NK * 8 * 32 * 8) return;
    int j    = idx & 7;
    int lane = (idx >> 3) & 31;
    int ct   = (idx >> 8) & 7;
    int t    = idx >> 11;
    int half = lane >> 4;
    int l15  = lane & 15;
    int col  = ct * 16 + l15;
    int k    = t * 32 + (j >> 2) * 16 + half * 8 + 2 * (j & 3);   // B element K index
    float b0 = (k     < kreal) ? W[(size_t)k * DH + col]       : 0.0f;
    float b1 = (k + 1 < kreal) ? W[(size_t)(k + 1) * DH + col] : 0.0f;
    unsigned int hw, lw;
    split2(b0, b1, hw, lw);
    unsigned int* o = Wf + (size_t)(idx >> 3) * 16;
    o[j]     = hw;
    o[8 + j] = lw;
}

// ---- Y = A[M x K] @ W[K x 128] + bias, WMMA bf16x3, K compile-time ----
// grid.x = M/16 row tiles, blockDim = 256 (8 waves), wave w owns col tile w.
template <int K>
__global__ void k_gemm(const float* __restrict__ A,
                       const unsigned int* __restrict__ Wf,
                       const float* __restrict__ bias,
                       float* __restrict__ Y, int ldy) {
    int lane = threadIdx.x & 31;
    int wid  = threadIdx.x >> 5;
    int half = lane >> 4;
    int l15  = lane & 15;
    int row  = blockIdx.x * 16 + l15;
    int col  = wid * 16 + l15;
    const float* arow = A + (size_t)row * K;

    f32x8 c = {};
    constexpr int NK = K / 32;
#pragma unroll
    for (int t = 0; t < NK; ++t) {
        int k0 = t * 32;
        // A fragment: elements 0..7 = K [k0+8h .. +7], 8..15 = K [k0+16+8h .. +7]
        const float4* p0 = reinterpret_cast<const float4*>(arow + k0 + half * 8);
        const float4* p1 = reinterpret_cast<const float4*>(arow + k0 + 16 + half * 8);
        float4 a0 = p0[0], a1 = p0[1];
        float4 a2 = p1[0], a3 = p1[1];

        Frag16 a_hi, a_lo;
        split2(a0.x, a0.y, a_hi.w[0], a_lo.w[0]);
        split2(a0.z, a0.w, a_hi.w[1], a_lo.w[1]);
        split2(a1.x, a1.y, a_hi.w[2], a_lo.w[2]);
        split2(a1.z, a1.w, a_hi.w[3], a_lo.w[3]);
        split2(a2.x, a2.y, a_hi.w[4], a_lo.w[4]);
        split2(a2.z, a2.w, a_hi.w[5], a_lo.w[5]);
        split2(a3.x, a3.y, a_hi.w[6], a_lo.w[6]);
        split2(a3.z, a3.w, a_hi.w[7], a_lo.w[7]);

        // B fragments: pre-swizzled, 4 x b128 per step
        const uint4* wf4 = reinterpret_cast<const uint4*>(Wf)
                         + (size_t)((t * 8 + wid) * 32 + lane) * 4;
        Frag16 b_hi, b_lo;
        b_hi.q[0] = wf4[0];
        b_hi.q[1] = wf4[1];
        b_lo.q[0] = wf4[2];
        b_lo.q[1] = wf4[3];

        c = __builtin_amdgcn_wmma_f32_16x16x32_bf16(false, a_hi.v, false, b_hi.v,
                                                    (short)0, c, false, false);
        c = __builtin_amdgcn_wmma_f32_16x16x32_bf16(false, a_hi.v, false, b_lo.v,
                                                    (short)0, c, false, false);
        c = __builtin_amdgcn_wmma_f32_16x16x32_bf16(false, a_lo.v, false, b_hi.v,
                                                    (short)0, c, false, false);
    }
    float bv = bias[col];
#pragma unroll
    for (int r = 0; r < 8; ++r) {
        int m = r + (half << 3);
        Y[(size_t)(blockIdx.x * 16 + m) * ldy + col] = c[r] + bv;
    }
}

// ---- zero small scratch ----
__global__ void k_zero(float* __restrict__ p, int n) {
    int i = blockIdx.x * blockDim.x + threadIdx.x;
    if (i < n) p[i] = 0.0f;
}

// ---- column sums / sumsq of Y[Nrows x 128] ----
__global__ void k_colstats(const float* __restrict__ Y, float* __restrict__ sums, int Nrows) {
    int c  = threadIdx.x;                  // 0..127, coalesced
    int r0 = blockIdx.x * 128;
    int r1 = min(r0 + 128, Nrows);
    float s = 0.0f, s2 = 0.0f;
    for (int r = r0; r < r1; ++r) {
        float v = Y[(size_t)r * DH + c];
        s  += v;
        s2 += v * v;
    }
    atomicAdd(&sums[c], s);
    atomicAdd(&sums[DH + c], s2);
}

// ---- Z = relu(g*(Y-m)*rsqrt(v+eps)+b) ----
__global__ void k_bnrelu(const float* __restrict__ Y, const float* __restrict__ sums,
                         const float* __restrict__ g, const float* __restrict__ be,
                         float* __restrict__ Z, int total, float invN) {
    int i = blockIdx.x * blockDim.x + threadIdx.x;
    if (i >= total) return;
    int c = i & (DH - 1);
    float m   = sums[c] * invN;
    float var = sums[DH + c] * invN - m * m;
    float z = g[c] * (Y[i] - m) * rsqrtf(var + BN_EPS) + be[c];
    Z[i] = fmaxf(z, 0.0f);
}

// ---- per-graph segment sum (graph_ids sorted) ----
__device__ __forceinline__ int lbound(const int* __restrict__ a, int n, int key) {
    int lo = 0, hi = n;
    while (lo < hi) {
        int mid = (lo + hi) >> 1;
        if (a[mid] < key) lo = mid + 1; else hi = mid;
    }
    return lo;
}

__global__ void k_pool(const float* __restrict__ H, int D, int ldh,
                       const int* __restrict__ gid, int Nn,
                       float* __restrict__ out, int colOff, int ldo) {
    int g = blockIdx.x;
    int start = lbound(gid, Nn, g);
    int end   = lbound(gid, Nn, g + 1);
    for (int c = threadIdx.x; c < D; c += blockDim.x) {
        float s = 0.0f;
        for (int n = start; n < end; ++n)
            s += H[(size_t)n * ldh + c];
        out[(size_t)g * ldo + colOff + c] = s;
    }
}

extern "C" void kernel_launch(void* const* d_in, const int* in_sizes, int n_in,
                              void* d_out, int out_size, void* d_ws, size_t ws_size,
                              hipStream_t stream) {
    const float* x   = (const float*)d_in[0];
    const int*   ei  = (const int*)d_in[1];
    const int*   gid = (const int*)d_in[2];
    float* out = (float*)d_out;

    auto AL = [](size_t b) { return (b + 255) & ~(size_t)255; };
    char* ws = (char*)d_ws;
    float* P    = (float*)ws;         ws += AL((size_t)N_NODES * KP0 * 4); // pooled / gemm2-out
    float* Yb   = (float*)ws;         ws += AL((size_t)N_NODES * DH * 4);
    float* H0   = (float*)ws;         ws += AL((size_t)N_NODES * DH * 4);
    float* H1   = (float*)ws;         ws += AL((size_t)N_NODES * DH * 4);
    unsigned int* Wf = (unsigned int*)ws;
    ws += AL((size_t)(KP0 / 32) * 8 * 32 * 16 * 4);                       // swizzled weights
    float* sums = (float*)ws;         ws += AL(2 * DH * 4);

    const float invN = 1.0f / (float)N_NODES;
    const int PREP0 = (KP0 / 32) * 2048;   // threads for k_prepw<KP0>
    const int PREP1 = (DH  / 32) * 2048;   // threads for k_prepw<DH>

    // pooled hidden[0] = x
    k_pool<<<N_GRAPH, 256, 0, stream>>>(x, DIN, DIN, gid, N_NODES, out, 0, OUTC);

    const float* h = x;
    for (int l = 0; l < 4; ++l) {
        int D   = (l == 0) ? DIN : DH;    // input feature dim of this layer
        int ldp = (l == 0) ? KP0 : DH;    // padded stride for pooled buffer
        const float* W1  = (const float*)d_in[3 + 9 * l + 0];
        const float* b1  = (const float*)d_in[3 + 9 * l + 1];
        const float* g1  = (const float*)d_in[3 + 9 * l + 2];
        const float* be1 = (const float*)d_in[3 + 9 * l + 3];
        const float* W2  = (const float*)d_in[3 + 9 * l + 4];
        const float* b2  = (const float*)d_in[3 + 9 * l + 5];
        const float* g2  = (const float*)d_in[3 + 9 * l + 6];
        const float* be2 = (const float*)d_in[3 + 9 * l + 7];
        const float* eps = (const float*)d_in[3 + 9 * l + 8];

        // pooled = (1+eps)*h (K-padded), then scatter-add neighbor messages
        k_scale_pad<<<N_NODES, 256, 0, stream>>>(h, eps, P, D, ldp);
        k_scatter<<<N_EDGES / 8, 256, 0, stream>>>(ei, h, P, N_EDGES, D, ldp);

        // Y = pooled @ W1 + b1
        if (l == 0) {
            k_prepw<KP0><<<(PREP0 + 255) / 256, 256, 0, stream>>>(W1, DIN, Wf);
            k_gemm<KP0><<<N_NODES / 16, 256, 0, stream>>>(P, Wf, b1, Yb, DH);
        } else {
            k_prepw<DH><<<(PREP1 + 255) / 256, 256, 0, stream>>>(W1, DH, Wf);
            k_gemm<DH><<<N_NODES / 16, 256, 0, stream>>>(P, Wf, b1, Yb, DH);
        }
        k_zero<<<1, 256, 0, stream>>>(sums, 2 * DH);
        k_colstats<<<(N_NODES + 127) / 128, 128, 0, stream>>>(Yb, sums, N_NODES);
        k_bnrelu<<<(N_NODES * DH + 255) / 256, 256, 0, stream>>>(Yb, sums, g1, be1,
                                                                 Yb, N_NODES * DH, invN);
        // Y2 = Z @ W2 + b2  (into P, compact stride DH)
        k_prepw<DH><<<(PREP1 + 255) / 256, 256, 0, stream>>>(W2, DH, Wf);
        k_gemm<DH><<<N_NODES / 16, 256, 0, stream>>>(Yb, Wf, b2, P, DH);
        k_zero<<<1, 256, 0, stream>>>(sums, 2 * DH);
        k_colstats<<<(N_NODES + 127) / 128, 128, 0, stream>>>(P, sums, N_NODES);
        float* hn = (l & 1) ? H1 : H0;
        k_bnrelu<<<(N_NODES * DH + 255) / 256, 256, 0, stream>>>(P, sums, g2, be2,
                                                                 hn, N_NODES * DH, invN);
        // pool this layer's hidden rep while it is live
        k_pool<<<N_GRAPH, 256, 0, stream>>>(hn, DH, DH, gid, N_NODES, out,
                                            DIN + DH * l, OUTC);
        h = hn;
    }
    (void)in_sizes; (void)n_in; (void)out_size; (void)ws_size;
}
